// ElmanRNN_64518998721183
// MI455X (gfx1250) — matrix operand order repaired and checked by reading, not compiled, not run on hardware
//
#include <hip/hip_runtime.h>

// Problem dims (compile-time constants from the reference)
#define B_ 512
#define T_ 256
#define I_ 2
#define H_ 2048
#define O_ 1

typedef __attribute__((ext_vector_type(16))) __bf16 v16bf;
typedef __attribute__((ext_vector_type(8)))  float  v8f;
typedef unsigned short ushort_t;

static __device__ __forceinline__ ushort_t f32_to_bf16(float f) {
    union { float f; unsigned u; } v; v.f = f;
    unsigned r = v.u + 0x7fffu + ((v.u >> 16) & 1u);   // round-to-nearest-even
    return (ushort_t)(r >> 16);
}
static __device__ __forceinline__ float bf16_to_f32(ushort_t u) {
    union { unsigned u; float f; } v; v.u = ((unsigned)u) << 16;
    return v.f;
}

// ---- CDNA5 async copy global->LDS (16B per lane), tracked by ASYNCcnt ------
static __device__ __forceinline__ void async_copy_b128(unsigned lds_byte_off,
                                                       const ushort_t* gsrc) {
    asm volatile("global_load_async_to_lds_b128 %0, %1, off"
                 :: "v"(lds_byte_off), "v"(gsrc) : "memory");
}
template <int N>
static __device__ __forceinline__ void wait_asynccnt() {
#if __has_builtin(__builtin_amdgcn_s_wait_asynccnt)
    __builtin_amdgcn_s_wait_asynccnt((short)N);
#else
    if (N == 0)      asm volatile("s_wait_asynccnt 0" ::: "memory");
    else if (N == 6) asm volatile("s_wait_asynccnt 6" ::: "memory");
    else             asm volatile("s_wait_asynccnt 0" ::: "memory");
#endif
}

// ---------------------------------------------------------------------------
// Kernel 1: pack Wh (= W_ih[:, I:]) into bf16 row-major [n][k]; zero h0.
// ---------------------------------------------------------------------------
__global__ void prep_kernel(const float* __restrict__ W_ih,
                            ushort_t* __restrict__ Wh_bf16,
                            ushort_t* __restrict__ h0) {
    const int stride = gridDim.x * blockDim.x;
    const int tid = blockIdx.x * blockDim.x + threadIdx.x;
    for (int i = tid; i < H_ * H_; i += stride) {
        int n = i >> 11;          // / H_
        int k = i & (H_ - 1);     // % H_
        Wh_bf16[i] = f32_to_bf16(W_ih[(size_t)n * (I_ + H_) + I_ + k]);
    }
    for (int i = tid; i < B_ * H_; i += stride) h0[i] = 0;
}

// ---------------------------------------------------------------------------
// Kernel 2: one recurrence step, LDS-staged double-buffered GEMM.
//   h_out[b,n] = tanh( xp(b,n,t) + sum_k h_in[b,k] * Wh[n,k] )
// Block tile: M=64 x N=128, 8 waves (4 M-waves x 2 N-halves), wave = 16x64.
// Per K=64 slice: stage A 64x64 (8KB) + B 128x64 (16KB) in LDS via async
// copy; double buffered (48KB LDS). 6 async b128 chunks per thread per slice.
// ---------------------------------------------------------------------------
#define KB_      64               // K per staged slice
#define A_US     (64 * KB_)       // A slice, ushorts (8KB)
#define B_US     (128 * KB_)      // B slice, ushorts (16KB)
#define NSLICE   (H_ / KB_)       // 32 slices

__global__ __launch_bounds__(256)
void step_kernel(const float* __restrict__ x,
                 const ushort_t* __restrict__ Wh,     // bf16 [H][H]
                 const float* __restrict__ W_ih,      // Wx = W_ih[:, 0:2]
                 const float* __restrict__ b_ih,
                 const ushort_t* __restrict__ h_in,   // bf16 [B][H]
                 ushort_t* __restrict__ h_out,        // bf16 [B][H]
                 int t) {
    __shared__ ushort_t Abuf[2][A_US];   // [m_local][k_local], 64-ushort rows
    __shared__ ushort_t Bbuf[2][B_US];   // [n_local][k_local], 64-ushort rows

    const int tid   = threadIdx.x;
    const int lane  = tid & 31;
    const int w     = tid >> 5;            // 0..7
    const int col   = lane & 15;
    const int half  = lane >> 4;

    const int mblk  = (int)blockIdx.x >> 4;       // 0..7
    const int nblk  = (int)blockIdx.x & 15;       // 0..15
    const int m0blk = mblk * 64;
    const int n0blk = nblk * 128;
    const int mw    = (w & 3) * 16;               // wave M offset in block
    const int nw    = (w >> 2) * 64;              // wave N offset in block

    const unsigned aBase = (unsigned)(uintptr_t)&Abuf[0][0];
    const unsigned bBase = (unsigned)(uintptr_t)&Bbuf[0][0];

    // Issue one K-slice's async batch: A 512 chunks (2/thread), B 1024 (4/thread)
    auto issue = [&](int bufsel, int kk) {
#pragma unroll
        for (int j = 0; j < 2; ++j) {
            const int c = tid + j * 256, row = c >> 3, seg = c & 7;
            async_copy_b128(aBase + (unsigned)(bufsel * A_US + row * KB_ + seg * 8) * 2u,
                            h_in + (size_t)(m0blk + row) * H_ + kk + seg * 8);
        }
#pragma unroll
        for (int j = 0; j < 4; ++j) {
            const int c = tid + j * 256, row = c >> 3, seg = c & 7;
            async_copy_b128(bBase + (unsigned)(bufsel * B_US + row * KB_ + seg * 8) * 2u,
                            Wh + (size_t)(n0blk + row) * H_ + kk + seg * 8);
        }
    };

    v8f acc[4] = {v8f{}, v8f{}, v8f{}, v8f{}};

    // Consume one staged K=64 slice: load ALL fragments, then 8 WMMAs so the
    // compiler can interleave matrix ops with partial s_wait_dscnt waits.
    auto consume = [&](int bufsel) {
        const ushort_t* As = &Abuf[bufsel][0];
        const ushort_t* Bs = &Bbuf[bufsel][0];
        v16bf a[2];
        v16bf b[2][4];
#pragma unroll
        for (int s = 0; s < 2; ++s) {
            const int aIdx = (mw + col) * KB_ + s * 32 + half * 8;
            struct { uint4 lo, hi; } ab;
            ab.lo = *(const uint4*)(As + aIdx);
            ab.hi = *(const uint4*)(As + aIdx + 16);
            a[s] = __builtin_bit_cast(v16bf, ab);
        }
#pragma unroll
        for (int s = 0; s < 2; ++s) {
#pragma unroll
            for (int tt = 0; tt < 4; ++tt) {
                const int bIdx = (nw + tt * 16 + col) * KB_ + s * 32 + half * 8;
                struct { uint4 lo, hi; } bb;
                bb.lo = *(const uint4*)(Bs + bIdx);
                bb.hi = *(const uint4*)(Bs + bIdx + 16);
                b[s][tt] = __builtin_bit_cast(v16bf, bb);
            }
        }
#pragma unroll
        for (int s = 0; s < 2; ++s)
#pragma unroll
            for (int tt = 0; tt < 4; ++tt)
                acc[tt] = __builtin_amdgcn_wmma_f32_16x16x32_bf16(
                    false, a[s], false, b[s][tt], (short)0, acc[tt], false, false);
    };

    // prologue: fill both buffers
    issue(0, 0);
    issue(1, KB_);

    // steady state: branch-free (slices 0 .. NSLICE-3)
    int buf = 0;
    for (int k0 = 0; k0 < H_ - 2 * KB_; k0 += KB_) {
        wait_asynccnt<6>();          // this slice landed; next slice may fly
        __syncthreads();
        consume(buf);
        __syncthreads();             // all waves done reading buf
        issue(buf, k0 + 2 * KB_);
        buf ^= 1;
    }
    // slice NSLICE-2: nothing more to issue
    wait_asynccnt<6>();
    __syncthreads();
    consume(buf);
    buf ^= 1;
    // slice NSLICE-1: last in flight
    wait_asynccnt<0>();
    __syncthreads();
    consume(buf);

    // Epilogue: fold in x-projection + bias, tanh, store bf16 for next step.
    // C/D layout: VGPR i holds row M = i + half*8, col N = col.
#pragma unroll
    for (int tt = 0; tt < 4; ++tt) {
        const int n = n0blk + nw + tt * 16 + col;
        const float* wrow = W_ih + (size_t)n * (I_ + H_);
        const float wx0 = wrow[0];
        const float wx1 = wrow[1];
        const float bi  = b_ih[n];
#pragma unroll
        for (int i = 0; i < 8; ++i) {
            const int b = m0blk + mw + half * 8 + i;
            const float* xp = x + ((size_t)b * T_ + t) * I_;
            float pre = acc[tt][i] + xp[0] * wx0 + xp[1] * wx1 + bi;
            h_out[(size_t)b * H_ + n] = f32_to_bf16(tanhf(pre));
        }
    }
}

// ---------------------------------------------------------------------------
// Kernel 3: out[b] = dot(h_final[b,:], W_ho[0,:]) + b_ho[0]   (O = 1)
// One wave per batch row, shuffle reduction (wave32).
// ---------------------------------------------------------------------------
__global__ __launch_bounds__(256)
void out_kernel(const ushort_t* __restrict__ h_final,
                const float* __restrict__ W_ho,
                const float* __restrict__ b_ho,
                float* __restrict__ out) {
    const int lane = threadIdx.x & 31;
    const int b = blockIdx.x * (blockDim.x >> 5) + (threadIdx.x >> 5);
    if (b >= B_) return;
    float s = 0.0f;
    const ushort_t* hr = h_final + (size_t)b * H_;
    for (int n = lane; n < H_; n += 32)
        s += bf16_to_f32(hr[n]) * W_ho[n];
#pragma unroll
    for (int off = 16; off > 0; off >>= 1)
        s += __shfl_xor(s, off, 32);
    if (lane == 0) out[b] = s + b_ho[0];
}

// ---------------------------------------------------------------------------
extern "C" void kernel_launch(void* const* d_in, const int* in_sizes, int n_in,
                              void* d_out, int out_size, void* d_ws, size_t ws_size,
                              hipStream_t stream) {
    const float* x    = (const float*)d_in[0];   // (B,T,I)
    const float* W_ih = (const float*)d_in[1];   // (H, I+H)
    const float* b_ih = (const float*)d_in[2];   // (H,)
    const float* W_ho = (const float*)d_in[3];   // (O, H)
    const float* b_ho = (const float*)d_in[4];   // (O,)
    float* out = (float*)d_out;                  // (B, O) = 512 floats

    // Workspace layout: Wh_bf16 (8 MB) | h buf0 (2 MB) | h buf1 (2 MB)
    char* ws = (char*)d_ws;
    ushort_t* Wh   = (ushort_t*)ws;
    ushort_t* buf0 = (ushort_t*)(ws + (size_t)H_ * H_ * sizeof(ushort_t));
    ushort_t* buf1 = buf0 + (size_t)B_ * H_;

    prep_kernel<<<1024, 256, 0, stream>>>(W_ih, Wh, buf0);

    ushort_t* hin  = buf0;
    ushort_t* hout = buf1;
    // Grid: (512/64) M-blocks x (2048/128) N-blocks = 128 blocks of 8 waves
    for (int t = 0; t < T_; ++t) {
        step_kernel<<<128, 256, 0, stream>>>(x, Wh, W_ih, b_ih, hin, hout, t);
        ushort_t* tmp = hin; hin = hout; hout = tmp;
    }
    // After an even number of steps, the final state is in `hin`.
    out_kernel<<<B_ / 8, 256, 0, stream>>>(hin, W_ho, b_ho, out);
}
